// GNN_63720134803555
// MI455X (gfx1250) — compile-verified, matrix operand order checked by reading, not run on hardware
//
#include <hip/hip_runtime.h>
#include <hip/hip_bf16.h>
#include <cstdint>
#include <cstddef>

#define HD __device__ __forceinline__

typedef __attribute__((ext_vector_type(16))) __bf16 v16bf;
typedef __attribute__((ext_vector_type(8)))  float  v8f;

static constexpr int H_  = 128;
static constexpr int IN_ = 768;

// ---- monotone float <-> uint encoding for integer atomic max/min on floats ----
HD unsigned fenc(float f) {
  unsigned u = __float_as_uint(f);
  return (u & 0x80000000u) ? ~u : (u | 0x80000000u);
}
HD float fdec(unsigned u) {
  unsigned v = (u & 0x80000000u) ? (u & 0x7FFFFFFFu) : ~u;
  return __uint_as_float(v);
}
static constexpr unsigned ENC_NEG_INF = 0x007FFFFFu;  // fenc(-inf)

// ---- weight prep: W [K][128] f32 (row-major) -> Wt [128][K] bf16 (transposed) ----
__global__ void k_wprep(const float* __restrict__ W, __bf16* __restrict__ Wt, int K) {
  int idx = blockIdx.x * blockDim.x + threadIdx.x;
  if (idx < K * H_) {
    int k = idx >> 7, c = idx & 127;
    Wt[(size_t)c * K + k] = (__bf16)W[idx];
  }
}

// ---- WMMA GEMM: C[n][128] = act(A[n][K]) @ Wt^T, act = relu if RELU_IN ----
// One wave computes a 16-row x 128-col strip: 8 col tiles, K/32 k-steps.
// A-frag layout (16-bit A 16x32): lane m=lane&15, kb=(lane&16)?8:0;
//   elems 0..7 -> K=kb..kb+7, elems 8..15 -> K=kb+16..kb+23.
// B-frag symmetric with N=lane&15. C/D: VGPR j -> M=j+((lane&16)?8:0), N=lane&15.
// Epilogue: stage 16x64 f32 half-tiles (two passes) in per-wave LDS (stride 68 to avoid bank
// conflicts), then 16 fully coalesced 512B global_store_b128 per wave.
template <int K, bool RELU_IN>
__global__ __launch_bounds__(256) void k_gemm(const float* __restrict__ A,
                                              const __bf16* __restrict__ Bt,
                                              float* __restrict__ C, int n) {
  __shared__ float lds[8][16 * 68];  // 4352B per wave, 34816B per block
  const int wid  = threadIdx.x >> 5;
  const int lane = threadIdx.x & 31;
  const int wave = blockIdx.x * 8 + wid;
  const int row0 = wave * 16;
  if (row0 >= n) return;  // wave-uniform: EXEC stays all-ones for WMMA
  const int m  = lane & 15;
  const int kb = (lane & 16) ? 8 : 0;
  int row = row0 + m;
  if (row >= n) row = n - 1;  // duplicate-clamp tail rows; stores handled below
  const float* __restrict__ arow = A + (size_t)row * K;

  v8f acc[8] = {};
  for (int k0 = 0; k0 < K; k0 += 32) {
    float4 a0 = *(const float4*)(arow + k0 + kb);
    float4 a1 = *(const float4*)(arow + k0 + kb + 4);
    float4 a2 = *(const float4*)(arow + k0 + kb + 16);
    float4 a3 = *(const float4*)(arow + k0 + kb + 20);
    if (RELU_IN) {
      a0.x = fmaxf(a0.x, 0.f); a0.y = fmaxf(a0.y, 0.f); a0.z = fmaxf(a0.z, 0.f); a0.w = fmaxf(a0.w, 0.f);
      a1.x = fmaxf(a1.x, 0.f); a1.y = fmaxf(a1.y, 0.f); a1.z = fmaxf(a1.z, 0.f); a1.w = fmaxf(a1.w, 0.f);
      a2.x = fmaxf(a2.x, 0.f); a2.y = fmaxf(a2.y, 0.f); a2.z = fmaxf(a2.z, 0.f); a2.w = fmaxf(a2.w, 0.f);
      a3.x = fmaxf(a3.x, 0.f); a3.y = fmaxf(a3.y, 0.f); a3.z = fmaxf(a3.z, 0.f); a3.w = fmaxf(a3.w, 0.f);
    }
    v16bf af;
    af[0]  = (__bf16)a0.x; af[1]  = (__bf16)a0.y; af[2]  = (__bf16)a0.z; af[3]  = (__bf16)a0.w;
    af[4]  = (__bf16)a1.x; af[5]  = (__bf16)a1.y; af[6]  = (__bf16)a1.z; af[7]  = (__bf16)a1.w;
    af[8]  = (__bf16)a2.x; af[9]  = (__bf16)a2.y; af[10] = (__bf16)a2.z; af[11] = (__bf16)a2.w;
    af[12] = (__bf16)a3.x; af[13] = (__bf16)a3.y; af[14] = (__bf16)a3.z; af[15] = (__bf16)a3.w;
#pragma unroll
    for (int t = 0; t < 8; t++) {
      const __bf16* bcol = Bt + (size_t)(t * 16 + m) * K + k0 + kb;
      union { v16bf v; uint4 u[2]; } bf;
      bf.u[0] = *(const uint4*)(bcol);
      bf.u[1] = *(const uint4*)(bcol + 16);
      acc[t] = __builtin_amdgcn_wmma_f32_16x16x32_bf16(
          false, af, false, bf.v, (short)0, acc[t], false, false);
    }
  }

  // ---- epilogue: two half-tile passes through LDS, coalesced b128 stores ----
  float* wl = lds[wid];
  const int  mb   = (lane & 16) ? 8 : 0;
  const bool full = (row0 + 16) <= n;
  const int  rr   = lane >> 4;          // 0 or 1: which of the 2 rows this lane copies
  const int  cg   = (lane & 15) * 4;    // float4 column group within 64-wide half
#pragma unroll
  for (int p = 0; p < 2; p++) {
#pragma unroll
    for (int tt = 0; tt < 4; tt++) {
      const int t = p * 4 + tt;
#pragma unroll
      for (int j = 0; j < 8; j++) wl[(mb + j) * 68 + tt * 16 + m] = acc[t][j];
    }
#pragma unroll
    for (int j2 = 0; j2 < 8; j2++) {
      const int r = j2 * 2 + rr;
      float4 v = *(const float4*)&wl[r * 68 + cg];
      if (full || (row0 + r) < n)
        *(float4*)&C[(size_t)(row0 + r) * H_ + p * 64 + cg] = v;
    }
  }
}

// ---- attention coefficients: as[n] = hp[n]·a_s, ad[n] = hp[n]·a_d (wave per node) ----
__global__ __launch_bounds__(256) void k_alpha(const float* __restrict__ hp,
                                               const float* __restrict__ a_s,
                                               const float* __restrict__ a_d,
                                               float* __restrict__ as_,
                                               float* __restrict__ ad_, int n) {
  const int node = blockIdx.x * 8 + (threadIdx.x >> 5);
  const int lane = threadIdx.x & 31;
  if (node >= n) return;
  float4 v = ((const float4*)(hp + (size_t)node * H_))[lane];
  float4 s = ((const float4*)a_s)[lane];
  float4 d = ((const float4*)a_d)[lane];
  float ss = v.x * s.x + v.y * s.y + v.z * s.z + v.w * s.w;
  float dd = v.x * d.x + v.y * d.y + v.z * d.z + v.w * d.w;
#pragma unroll
  for (int o = 16; o; o >>= 1) {
    ss += __shfl_down(ss, o, 32);
    dd += __shfl_down(dd, o, 32);
  }
  if (lane == 0) { as_[node] = ss; ad_[node] = dd; }
}

// ---- per-layer node-state init: acc = bias, segment-max = -inf, denom = 0 ----
__global__ void k_init_nodes(float* __restrict__ acc, const float* __restrict__ b,
                             unsigned* __restrict__ menc, float* __restrict__ denom, int n) {
  int idx = blockIdx.x * blockDim.x + threadIdx.x;
  if (idx < n * H_) {
    int c = idx & 127;
    acc[idx] = b[c];
    if (c == 0) { int node = idx >> 7; menc[node] = ENC_NEG_INF; denom[node] = 0.f; }
  }
}

HD void edge_sd(const int* __restrict__ ei, int E, int e, int& s, int& d) {
  if (e < E) { s = ei[e]; d = ei[E + e]; } else { s = d = e - E; }  // self-loops appended
}

HD float lrelu02(float x) { return x > 0.f ? x : 0.2f * x; }

// ---- edge pass 1: segment max of leaky-relu logits over dst ----
__global__ void k_edge_max(const int* __restrict__ ei, int E, int n,
                           const float* __restrict__ as_, const float* __restrict__ ad_,
                           unsigned* __restrict__ menc) {
  int e = blockIdx.x * blockDim.x + threadIdx.x;
  if (e >= E + n) return;
  int s, d; edge_sd(ei, E, e, s, d);
  atomicMax(&menc[d], fenc(lrelu02(as_[s] + ad_[d])));
}

// ---- edge pass 2: ex = exp(logit - max), segment-sum denom ----
__global__ void k_edge_exp(const int* __restrict__ ei, int E, int n,
                           const float* __restrict__ as_, const float* __restrict__ ad_,
                           const unsigned* __restrict__ menc, float* __restrict__ exbuf,
                           float* __restrict__ denom) {
  int e = blockIdx.x * blockDim.x + threadIdx.x;
  if (e >= E + n) return;
  int s, d; edge_sd(ei, E, e, s, d);
  float lg = lrelu02(as_[s] + ad_[d]);
  float ex = __expf(lg - fdec(menc[d]));
  exbuf[e] = ex;
  atomicAdd(&denom[d], ex);
}

// ---- edge pass 3: acc[dst] += hp[src] * (ex/denom[dst]); one wave per edge ----
__global__ __launch_bounds__(256) void k_edge_scatter(const int* __restrict__ ei, int E, int n,
                                                      const float* __restrict__ hp,
                                                      const float* __restrict__ exbuf,
                                                      const float* __restrict__ denom,
                                                      float* __restrict__ acc) {
  int e = blockIdx.x * 8 + (threadIdx.x >> 5);
  int lane = threadIdx.x & 31;
  if (e >= E + n) return;
  int s, d; edge_sd(ei, E, e, s, d);
  float w = exbuf[e] / denom[d];
  float4 v = ((const float4*)(hp + (size_t)s * H_))[lane];
  float* dst = acc + (size_t)d * H_ + lane * 4;
  atomicAdd(dst + 0, v.x * w);
  atomicAdd(dst + 1, v.y * w);
  atomicAdd(dst + 2, v.z * w);
  atomicAdd(dst + 3, v.w * w);
}

// ---- pooling / root ----
__global__ void k_pool_init(unsigned* __restrict__ hg, unsigned* __restrict__ root, int G) {
  int i = blockIdx.x * blockDim.x + threadIdx.x;
  if (i < G * H_) hg[i] = ENC_NEG_INF;
  if (i < G) root[i] = 0x7FFFFFFFu;
}
__global__ void k_root(const int* __restrict__ batch, unsigned* __restrict__ root, int n) {
  int i = blockIdx.x * blockDim.x + threadIdx.x;
  if (i < n) atomicMin(&root[batch[i]], (unsigned)i);
}
// fuses the layer-3 relu: pools relu(h) per graph
__global__ void k_pool(const float* __restrict__ h, const int* __restrict__ batch,
                       unsigned* __restrict__ hg, int n) {
  int idx = blockIdx.x * blockDim.x + threadIdx.x;
  if (idx >= n * H_) return;
  int node = idx >> 7, c = idx & 127;
  atomicMax(&hg[batch[node] * H_ + c], fenc(fmaxf(h[idx], 0.f)));
}

// ---- head: relu(hg@W0+b0), relu(x[root]@Wnews+bnews), sigmoid(concat @ Wout + bout) ----
__global__ __launch_bounds__(128) void k_head(const unsigned* __restrict__ hg_enc,
                                              const unsigned* __restrict__ root,
                                              const float* __restrict__ x,
                                              const float* __restrict__ W0,
                                              const float* __restrict__ b0,
                                              const float* __restrict__ Wnews,
                                              const float* __restrict__ bnews,
                                              const float* __restrict__ Wout,
                                              const float* __restrict__ bout,
                                              float* __restrict__ out) {
  __shared__ float hgs[H_];
  __shared__ float red[H_];
  int g = blockIdx.x, c = threadIdx.x;
  hgs[c] = fdec(hg_enc[g * H_ + c]);
  __syncthreads();
  float t1 = b0[c];
#pragma unroll 4
  for (int k = 0; k < H_; k++) t1 += hgs[k] * W0[k * H_ + c];
  t1 = fmaxf(t1, 0.f);
  const float* xr = x + (size_t)root[g] * IN_;
  float t2 = bnews[c];
  for (int k = 0; k < IN_; k++) t2 += xr[k] * Wnews[k * H_ + c];
  t2 = fmaxf(t2, 0.f);
  red[c] = t1 * Wout[c] + t2 * Wout[H_ + c];
  __syncthreads();
  for (int off = 64; off > 0; off >>= 1) {
    if (c < off) red[c] += red[c + off];
    __syncthreads();
  }
  if (c == 0) {
    float z = red[0] + bout[0];
    out[g] = 1.f / (1.f + __expf(-z));
  }
}

extern "C" void kernel_launch(void* const* d_in, const int* in_sizes, int n_in,
                              void* d_out, int out_size, void* d_ws, size_t ws_size,
                              hipStream_t stream) {
  const float* x     = (const float*)d_in[0];
  const int*   ei    = (const int*)d_in[1];
  const int*   batch = (const int*)d_in[2];
  const float* W1    = (const float*)d_in[4];
  const float* as1   = (const float*)d_in[5];
  const float* ad1   = (const float*)d_in[6];
  const float* b1    = (const float*)d_in[7];
  const float* W2    = (const float*)d_in[8];
  const float* as2   = (const float*)d_in[9];
  const float* ad2   = (const float*)d_in[10];
  const float* b2    = (const float*)d_in[11];
  const float* W3    = (const float*)d_in[12];
  const float* as3   = (const float*)d_in[13];
  const float* ad3   = (const float*)d_in[14];
  const float* b3    = (const float*)d_in[15];
  const float* Wnews = (const float*)d_in[16];
  const float* bnews = (const float*)d_in[17];
  const float* W0    = (const float*)d_in[18];
  const float* b0    = (const float*)d_in[19];
  const float* Wout  = (const float*)d_in[20];
  const float* bout  = (const float*)d_in[21];
  float* out = (float*)d_out;

  const int N = in_sizes[0] / IN_;
  const int E = in_sizes[1] / 2;
  const int G = out_size;  // output is [G,1]
  const int etot = E + N;

  // workspace carve-out
  char* wsb = (char*)d_ws;
  size_t off = 0;
  auto take = [&](size_t bytes) -> char* {
    char* p = wsb + off;
    off = (off + bytes + 255) & ~(size_t)255;
    return p;
  };
  float*    hp    = (float*)take((size_t)N * H_ * 4);
  float*    acc   = (float*)take((size_t)N * H_ * 4);
  float*    as_   = (float*)take((size_t)N * 4);
  float*    ad_   = (float*)take((size_t)N * 4);
  unsigned* menc  = (unsigned*)take((size_t)N * 4);
  float*    denom = (float*)take((size_t)N * 4);
  float*    exbuf = (float*)take((size_t)etot * 4);
  __bf16*   Wt1   = (__bf16*)take((size_t)IN_ * H_ * 2);
  __bf16*   Wt2   = (__bf16*)take((size_t)H_ * H_ * 2);
  __bf16*   Wt3   = (__bf16*)take((size_t)H_ * H_ * 2);
  unsigned* hg    = (unsigned*)take((size_t)G * H_ * 4);
  unsigned* root  = (unsigned*)take((size_t)G * 4);
  (void)n_in; (void)ws_size;

  const int NH = N * H_;
  const int gblocks = (((N + 15) / 16) + 7) / 8;  // 16-row strip per wave, 8 waves/block

  // weight prep (tiny; stays resident in L2)
  k_wprep<<<(IN_ * H_ + 255) / 256, 256, 0, stream>>>(W1, Wt1, IN_);
  k_wprep<<<(H_ * H_ + 255) / 256, 256, 0, stream>>>(W2, Wt2, H_);
  k_wprep<<<(H_ * H_ + 255) / 256, 256, 0, stream>>>(W3, Wt3, H_);

  // graph-level init + root extraction (independent of layers)
  k_pool_init<<<(G * H_ + 255) / 256, 256, 0, stream>>>(hg, root, G);
  k_root<<<(N + 255) / 256, 256, 0, stream>>>(batch, root, N);

  // ---- layer 1 (input x, K=768) ----
  k_gemm<IN_, false><<<gblocks, 256, 0, stream>>>(x, Wt1, hp, N);
  k_alpha<<<(N + 7) / 8, 256, 0, stream>>>(hp, as1, ad1, as_, ad_, N);
  k_init_nodes<<<(NH + 255) / 256, 256, 0, stream>>>(acc, b1, menc, denom, N);
  k_edge_max<<<(etot + 255) / 256, 256, 0, stream>>>(ei, E, N, as_, ad_, menc);
  k_edge_exp<<<(etot + 255) / 256, 256, 0, stream>>>(ei, E, N, as_, ad_, menc, exbuf, denom);
  k_edge_scatter<<<(etot + 7) / 8, 256, 0, stream>>>(ei, E, N, hp, exbuf, denom, acc);

  // ---- layer 2 (input relu(acc)=h1, K=128; relu fused into A-read) ----
  k_gemm<H_, true><<<gblocks, 256, 0, stream>>>(acc, Wt2, hp, N);
  k_alpha<<<(N + 7) / 8, 256, 0, stream>>>(hp, as2, ad2, as_, ad_, N);
  k_init_nodes<<<(NH + 255) / 256, 256, 0, stream>>>(acc, b2, menc, denom, N);
  k_edge_max<<<(etot + 255) / 256, 256, 0, stream>>>(ei, E, N, as_, ad_, menc);
  k_edge_exp<<<(etot + 255) / 256, 256, 0, stream>>>(ei, E, N, as_, ad_, menc, exbuf, denom);
  k_edge_scatter<<<(etot + 7) / 8, 256, 0, stream>>>(ei, E, N, hp, exbuf, denom, acc);

  // ---- layer 3 (input relu(acc)=h2, K=128; relu fused into A-read) ----
  k_gemm<H_, true><<<gblocks, 256, 0, stream>>>(acc, Wt3, hp, N);
  k_alpha<<<(N + 7) / 8, 256, 0, stream>>>(hp, as3, ad3, as_, ad_, N);
  k_init_nodes<<<(NH + 255) / 256, 256, 0, stream>>>(acc, b3, menc, denom, N);
  k_edge_max<<<(etot + 255) / 256, 256, 0, stream>>>(ei, E, N, as_, ad_, menc);
  k_edge_exp<<<(etot + 255) / 256, 256, 0, stream>>>(ei, E, N, as_, ad_, menc, exbuf, denom);
  k_edge_scatter<<<(etot + 7) / 8, 256, 0, stream>>>(ei, E, N, hp, exbuf, denom, acc);

  // ---- pooling (fuses layer-3 relu) + head ----
  k_pool<<<(NH + 255) / 256, 256, 0, stream>>>(acc, batch, hg, N);
  k_head<<<G, 128, 0, stream>>>(hg, root, x, W0, b0, Wnews, bnews, Wout, bout, out);
}
